// BitLinear_18708877541742
// MI455X (gfx1250) — compile-verified, matrix operand order checked by reading, not run on hardware
//
#include <hip/hip_runtime.h>
#include <hip/hip_bf16.h>

// ---------------------------------------------------------------------------
// BitLinear forward for MI455X (gfx1250, wave32, WMMA).
//   X:[4,2048,4096] f32, W:[16384,4096] f32 -> out:[4,2048,16384] f32
//   out = (X @ Wq^T) * scale,  Wq = clip(round(W/scale)) in {-1,0,1}
// Wq exactly representable in bf16 -> bf16 WMMA, f32 accumulate.
// GEMM staging: CDNA5 async global->LDS (ASYNCcnt) + double buffering,
// last K-iteration peeled so the steady-state loop is branch-free.
// ---------------------------------------------------------------------------

typedef __attribute__((ext_vector_type(16))) __bf16 v16bf;
typedef __attribute__((ext_vector_type(8)))  float  v8f;

#define GEMM_M 8192
#define GEMM_N 16384
#define GEMM_K 4096
#define BM 128
#define BN 128
#define BK 32
#define RED_BLOCKS 1024

__device__ __forceinline__ __bf16 f2bf(float f) {
    union { float f; unsigned u; } v; v.f = f;
    unsigned r = v.u + 0x7FFFu + ((v.u >> 16) & 1u);   // round-to-nearest-even
    unsigned short h = (unsigned short)(r >> 16);
    return __builtin_bit_cast(__bf16, h);
}

// Async copy 32B (2 x b128) global -> LDS; tracked by ASYNCcnt, no VGPR data.
__device__ __forceinline__ void async_ld32(unsigned lds_off, unsigned long long gaddr) {
    asm volatile("global_load_async_to_lds_b128 %0, %1, off\n\t"
                 "global_load_async_to_lds_b128 %0, %1, off offset:16"
                 :: "v"(lds_off), "v"(gaddr) : "memory");
}
#define WAIT_ASYNC_LE4() asm volatile("s_wait_asynccnt 4" ::: "memory")
#define WAIT_ASYNC_0()   asm volatile("s_wait_asynccnt 0" ::: "memory")

// ---- Pass 1: per-block partial sums of |W| (deterministic 2-pass reduce) ----
__global__ __launch_bounds__(256) void k_abs_partial(const float* __restrict__ w,
                                                     float* __restrict__ partials,
                                                     long n4) {
    __shared__ float sm[256];
    const float4* w4 = (const float4*)w;
    long stride = (long)gridDim.x * 256;
    float s = 0.f;
    for (long i = (long)blockIdx.x * 256 + threadIdx.x; i < n4; i += stride) {
        float4 v = w4[i];
        s += fabsf(v.x) + fabsf(v.y) + fabsf(v.z) + fabsf(v.w);
    }
    sm[threadIdx.x] = s;
    __syncthreads();
    for (int o = 128; o > 0; o >>= 1) {
        if ((int)threadIdx.x < o) sm[threadIdx.x] += sm[threadIdx.x + o];
        __syncthreads();
    }
    if (threadIdx.x == 0) partials[blockIdx.x] = sm[0];
}

// ---- Pass 2: finalize scale = max(mean|W|, 1e-5); also store 1/scale ----
__global__ __launch_bounds__(256) void k_scale(const float* __restrict__ partials,
                                               float* __restrict__ scalars) {
    __shared__ float sm[256];
    float s = partials[threadIdx.x] + partials[threadIdx.x + 256] +
              partials[threadIdx.x + 512] + partials[threadIdx.x + 768];
    sm[threadIdx.x] = s;
    __syncthreads();
    for (int o = 128; o > 0; o >>= 1) {
        if ((int)threadIdx.x < o) sm[threadIdx.x] += sm[threadIdx.x + o];
        __syncthreads();
    }
    if (threadIdx.x == 0) {
        float mean = sm[0] / (float)((long)GEMM_N * GEMM_K);
        float scale = fmaxf(mean, 1e-5f);
        scalars[0] = scale;
        scalars[1] = 1.0f / scale;
    }
}

// ---- Pass 3: quantize W -> ternary bf16 (8 elems / thread) ----
__global__ __launch_bounds__(256) void k_quant_w(const float* __restrict__ w,
                                                 __bf16* __restrict__ wq,
                                                 const float* __restrict__ scalars) {
    const float inv = scalars[1];
    long base = ((long)blockIdx.x * 256 + threadIdx.x) * 8;
    const float4* w4 = (const float4*)(w + base);
    float4 a = w4[0], b = w4[1];
    __bf16 q[8];
    float v[8] = {a.x, a.y, a.z, a.w, b.x, b.y, b.z, b.w};
#pragma unroll
    for (int i = 0; i < 8; ++i) {
        float t = fminf(fmaxf(rintf(v[i] * inv), -1.0f), 1.0f);
        q[i] = f2bf(t);
    }
    *(uint4*)(wq + base) = *(const uint4*)q;
}

// ---- Pass 4: convert X -> bf16 (8 elems / thread) ----
__global__ __launch_bounds__(256) void k_cvt_x(const float* __restrict__ x,
                                               __bf16* __restrict__ xb) {
    long base = ((long)blockIdx.x * 256 + threadIdx.x) * 8;
    const float4* x4 = (const float4*)(x + base);
    float4 a = x4[0], b = x4[1];
    float v[8] = {a.x, a.y, a.z, a.w, b.x, b.y, b.z, b.w};
    __bf16 q[8];
#pragma unroll
    for (int i = 0; i < 8; ++i) q[i] = f2bf(v[i]);
    *(uint4*)(xb + base) = *(const uint4*)q;
}

// ---- Pass 5: bf16 WMMA GEMM: out[m,n] = scale * sum_k xb[m,k]*wq[n,k] ----
// Block tile 128x128, BK=32, double-buffered async LDS staging.
// 8 waves: wave (wm=w/2, wn=w%2) owns 32(M)x64(N) -> 2x4 WMMA accumulators.
__global__ __launch_bounds__(256) void k_gemm(const __bf16* __restrict__ A,  // [M,K] row-major
                                              const __bf16* __restrict__ B,  // [N,K] row-major
                                              const float* __restrict__ scalars,
                                              float* __restrict__ C) {
    __shared__ alignas(32) __bf16 As[2][BM * BK];   // [buf][m][k], 2 x 8 KB
    __shared__ alignas(32) __bf16 Bs[2][BN * BK];   // [buf][n][k], 2 x 8 KB

    const int tid  = threadIdx.x;
    const int lane = tid & 31;
    const int wave = tid >> 5;
    const int wm   = wave >> 1;          // 0..3
    const int wn   = wave & 1;           // 0..1
    const int m0   = blockIdx.y * BM;
    const int n0   = blockIdx.x * BN;
    const float scale = scalars[0];

    // staging assignment: 2 threads per row, 32B (16 bf16) each
    const int row = tid >> 1;            // 0..127
    const int seg = (tid & 1) * 16;      // element offset 0 or 16 within BK
    const __bf16* gA = A + (long)(m0 + row) * GEMM_K + seg;
    const __bf16* gB = B + (long)(n0 + row) * GEMM_K + seg;

    // per-thread LDS byte offsets (low 32 bits of generic shared address)
    const unsigned ldsA[2] = { (unsigned)(uintptr_t)&As[0][row * BK + seg],
                               (unsigned)(uintptr_t)&As[1][row * BK + seg] };
    const unsigned ldsB[2] = { (unsigned)(uintptr_t)&Bs[0][row * BK + seg],
                               (unsigned)(uintptr_t)&Bs[1][row * BK + seg] };

    // fragment read coords (A 16x32 / B 32x16 in one contiguous k order for both, not the ISA table order):
    // lane L: row/col = L%16, K half = (L/16)*16 ; 16 contiguous bf16 = 32B
    const int frow = lane & 15;
    const int kp   = (lane >> 4) * 16;

    v8f acc[2][4] = {};

    auto compute_tile = [&](int buf) __attribute__((always_inline)) {
        v16bf afrag[2], bfrag[4];
#pragma unroll
        for (int mt = 0; mt < 2; ++mt)
            afrag[mt] = *(const v16bf*)(&As[buf][(wm * 32 + mt * 16 + frow) * BK + kp]);
#pragma unroll
        for (int nt = 0; nt < 4; ++nt)
            bfrag[nt] = *(const v16bf*)(&Bs[buf][(wn * 64 + nt * 16 + frow) * BK + kp]);
#pragma unroll
        for (int mt = 0; mt < 2; ++mt)
#pragma unroll
            for (int nt = 0; nt < 4; ++nt)
                acc[mt][nt] = __builtin_amdgcn_wmma_f32_16x16x32_bf16(
                    /*neg_a=*/false, afrag[mt],
                    /*neg_b=*/false, bfrag[nt],
                    /*c_mod=*/(short)0, acc[mt][nt],
                    /*reuse_a=*/false, /*reuse_b=*/false);
    };

    // prologue: stage tile 0 into buffer 0 (4 async b128 per thread)
    async_ld32(ldsA[0], (unsigned long long)(uintptr_t)(gA));
    async_ld32(ldsB[0], (unsigned long long)(uintptr_t)(gB));

    // steady state (last iteration peeled): branch-free body
    int buf = 0;
    for (int k0 = 0; k0 < GEMM_K - BK; k0 += BK) {
        async_ld32(ldsA[buf ^ 1], (unsigned long long)(uintptr_t)(gA + k0 + BK));
        async_ld32(ldsB[buf ^ 1], (unsigned long long)(uintptr_t)(gB + k0 + BK));
        __builtin_prefetch((const void*)(gA + k0 + 2 * BK), 0, 1);  // warm L2 ahead
        __builtin_prefetch((const void*)(gB + k0 + 2 * BK), 0, 1);
        WAIT_ASYNC_LE4();   // current tile's 4 async loads complete (in-order)
        __syncthreads();    // tile visible to all waves
        compute_tile(buf);
        __syncthreads();    // all reads of buf done before it is re-staged
        buf ^= 1;
    }
    // final tile
    WAIT_ASYNC_0();
    __syncthreads();
    compute_tile(buf);

    // epilogue: C/D layout -> VGPR r holds M = base + 8*(lane/16) + r, N = base + lane%16
    const int mlo = (lane >> 4) * 8;
    const int nn  = lane & 15;
#pragma unroll
    for (int mt = 0; mt < 2; ++mt) {
#pragma unroll
        for (int nt = 0; nt < 4; ++nt) {
            long mbase = m0 + wm * 32 + mt * 16 + mlo;
            long nbase = n0 + wn * 64 + nt * 16 + nn;
#pragma unroll
            for (int r = 0; r < 8; ++r)
                C[(mbase + r) * (long)GEMM_N + nbase] = acc[mt][nt][r] * scale;
        }
    }
}

// ---------------------------------------------------------------------------
extern "C" void kernel_launch(void* const* d_in, const int* in_sizes, int n_in,
                              void* d_out, int out_size, void* d_ws, size_t ws_size,
                              hipStream_t stream) {
    const float* x = (const float*)d_in[0];   // 4*2048*4096 f32
    const float* w = (const float*)d_in[1];   // 16384*4096 f32
    float* out = (float*)d_out;               // 4*2048*16384 f32

    const size_t WQ_BYTES = (size_t)GEMM_N * GEMM_K * 2;   // 134,217,728
    const size_t XB_BYTES = (size_t)GEMM_M * GEMM_K * 2;   //  67,108,864
    char* ws = (char*)d_ws;
    __bf16* wq       = (__bf16*)ws;
    __bf16* xb       = (__bf16*)(ws + WQ_BYTES);
    float*  partials = (float*)(ws + WQ_BYTES + XB_BYTES);
    float*  scalars  = partials + RED_BLOCKS;

    const long wn4 = (long)GEMM_N * GEMM_K / 4;

    k_abs_partial<<<RED_BLOCKS, 256, 0, stream>>>(w, partials, wn4);
    k_scale<<<1, 256, 0, stream>>>(partials, scalars);

    const int qw_blocks = (int)((long)GEMM_N * GEMM_K / (256 * 8));  // 32768
    k_quant_w<<<qw_blocks, 256, 0, stream>>>(w, wq, scalars);

    const int cx_blocks = (int)((long)GEMM_M * GEMM_K / (256 * 8));  // 16384
    k_cvt_x<<<cx_blocks, 256, 0, stream>>>(x, xb);

    dim3 grid(GEMM_N / BN, GEMM_M / BM);   // 128 x 64
    k_gemm<<<grid, 256, 0, stream>>>(xb, wq, scalars, out);
}
